// ArchSampler_35931696398525
// MI455X (gfx1250) — matrix operand order repaired and checked by reading, not compiled, not run on hardware
//
#include <hip/hip_runtime.h>
#include <hip/hip_bf16.h>

#define HS    2048
#define G4H   8192           // 4*HS
#define NLAY  2
#define STEPS 64
#define TYPES 32

typedef __bf16 bf16_t;
typedef __attribute__((ext_vector_type(16))) __bf16 v16bf;
typedef __attribute__((ext_vector_type(8)))  __bf16 v8bf;
typedef __attribute__((ext_vector_type(8)))  float  v8f;

__device__ __forceinline__ bf16_t f2bf(float f) {
  unsigned int u = __builtin_bit_cast(unsigned int, f);
  u += 0x7FFFu + ((u >> 16) & 1u);                 // round-to-nearest-even
  unsigned short s = (unsigned short)(u >> 16);
  return __builtin_bit_cast(bf16_t, s);
}

// ---------------- one-time fp32 -> bf16 weight conversion ----------------
__global__ void cvt_f32_to_bf16(const float* __restrict__ src,
                                bf16_t* __restrict__ dst, int n) {
  int i = blockIdx.x * blockDim.x + threadIdx.x;
  int stride = gridDim.x * blockDim.x;
  for (; i < n; i += stride) dst[i] = f2bf(src[i]);
}

// ---------------- init h/c/x state ----------------
__global__ void init_state(const float* __restrict__ g_emb, bf16_t* __restrict__ x_bf,
                           bf16_t* __restrict__ h0_bf, bf16_t* __restrict__ h1_bf,
                           float* __restrict__ c0, float* __restrict__ c1) {
  int i = blockIdx.x * blockDim.x + threadIdx.x;
  if (i < HS) {
    x_bf[i]  = f2bf(g_emb[i]);
    h0_bf[i] = f2bf(0.f);
    h1_bf[i] = f2bf(0.f);
    c0[i] = 0.f;
    c1[i] = 0.f;
  }
}

// ---------------- gates = Wih @ x + Whh @ h  via bf16 WMMA ----------------
// One wave owns 16 output rows; B is the input vector broadcast across all
// 16 N-columns, so every column of D equals the matvec result.
__global__ __launch_bounds__(128) void lstm_matvec(
    const bf16_t* __restrict__ Wih, const bf16_t* __restrict__ Whh,
    const bf16_t* __restrict__ xin, const bf16_t* __restrict__ hin,
    float* __restrict__ gates) {
  const int lane  = threadIdx.x & 31;
  const int wave  = threadIdx.x >> 5;
  const int tile  = blockIdx.x * 4 + wave;        // 512 tiles of 16 rows
  const int m0    = tile * 16;
  const int laneM = lane & 15;
  const int hi    = lane >> 4;                    // 0: K 0..7/16..23, 1: K 8..15/24..31

  v8f acc = {};
  const bf16_t* rowI = Wih + (size_t)(m0 + laneM) * HS;
  const bf16_t* rowH = Whh + (size_t)(m0 + laneM) * HS;

#pragma unroll 4
  for (int k0 = 0; k0 < HS; k0 += 32) {
    v16bf a, b;
    v8bf lo = *(const v8bf*)(rowI + k0 + hi * 8);
    v8bf hh = *(const v8bf*)(rowI + k0 + 16 + hi * 8);
#pragma unroll
    for (int i = 0; i < 8; ++i) { a[i] = lo[i]; a[8 + i] = hh[i]; }
    v8bf blo = *(const v8bf*)(xin + k0 + hi * 8);
    v8bf bhh = *(const v8bf*)(xin + k0 + 16 + hi * 8);
#pragma unroll
    for (int i = 0; i < 8; ++i) { b[i] = blo[i]; b[8 + i] = bhh[i]; }
    acc = __builtin_amdgcn_wmma_f32_16x16x32_bf16(false, a, false, b,
                                                  (short)0, acc, false, false);
  }
#pragma unroll 4
  for (int k0 = 0; k0 < HS; k0 += 32) {
    v16bf a, b;
    v8bf lo = *(const v8bf*)(rowH + k0 + hi * 8);
    v8bf hh = *(const v8bf*)(rowH + k0 + 16 + hi * 8);
#pragma unroll
    for (int i = 0; i < 8; ++i) { a[i] = lo[i]; a[8 + i] = hh[i]; }
    v8bf blo = *(const v8bf*)(hin + k0 + hi * 8);
    v8bf bhh = *(const v8bf*)(hin + k0 + 16 + hi * 8);
#pragma unroll
    for (int i = 0; i < 8; ++i) { b[i] = blo[i]; b[8 + i] = bhh[i]; }
    acc = __builtin_amdgcn_wmma_f32_16x16x32_bf16(false, a, false, b,
                                                  (short)0, acc, false, false);
  }

  // D layout: vgpr r, lanes 0-15 -> (M=r, N=lane); lanes 16-31 -> (M=8+r).
  // All columns identical; column 0 lives in lane 0 (rows 0..7) and lane 16 (rows 8..15).
  if (lane == 0) {
#pragma unroll
    for (int r = 0; r < 8; ++r) gates[m0 + r] = acc[r];
  } else if (lane == 16) {
#pragma unroll
    for (int r = 0; r < 8; ++r) gates[m0 + 8 + r] = acc[r];
  }
}

// ---------------- LSTM cell: i,f,g,o -> h,c ----------------
__global__ void lstm_cell(const float* __restrict__ gates, float* __restrict__ c,
                          bf16_t* __restrict__ h_bf, float* __restrict__ h_f32) {
  int j = blockIdx.x * blockDim.x + threadIdx.x;
  if (j >= HS) return;
  float ig = gates[j];
  float fg = gates[HS + j];
  float gg = gates[2 * HS + j];
  float og = gates[3 * HS + j];
  float si = 1.f / (1.f + __expf(-ig));
  float sf = 1.f / (1.f + __expf(-fg));
  float so = 1.f / (1.f + __expf(-og));
  float cn = sf * c[j] + si * tanhf(gg);
  float hn = so * tanhf(cn);
  c[j]    = cn;
  h_bf[j] = f2bf(hn);
  h_f32[j] = hn;
}

// ---------------- logits, Gumbel-argmax, log-softmax, next-x gather ----------------
__global__ __launch_bounds__(256) void sampler(
    const float* __restrict__ h1, const float* __restrict__ w_soft,
    const float* __restrict__ gumbel, const float* __restrict__ w_emb,
    bf16_t* __restrict__ x_bf, float* __restrict__ out, int t) {
  __shared__ float part[8][TYPES];
  __shared__ int s_idx;
  const int tid = threadIdx.x;
  const int col = tid & 31;
  const int p   = tid >> 5;

  float s = 0.f;
  for (int k = p * 256; k < (p + 1) * 256; ++k)
    s += h1[k] * w_soft[k * TYPES + col];
  part[p][col] = s;
  __syncthreads();

  if (tid < TYPES) {                       // wave 0 (wave32) does the reductions
    float logit = 0.f;
#pragma unroll
    for (int q = 0; q < 8; ++q) logit += part[q][col];

    float pert = logit + gumbel[t * TYPES + col];
    float bm = pert; int bi = col;
#pragma unroll
    for (int off = 16; off > 0; off >>= 1) {
      float om = __shfl_xor(bm, off, 32);
      int   ob = __shfl_xor(bi, off, 32);
      if (om > bm || (om == bm && ob < bi)) { bm = om; bi = ob; }
    }
    float ml = logit;
#pragma unroll
    for (int off = 16; off > 0; off >>= 1) ml = fmaxf(ml, __shfl_xor(ml, off, 32));
    float e = __expf(logit - ml);
#pragma unroll
    for (int off = 16; off > 0; off >>= 1) e += __shfl_xor(e, off, 32);

    float logit_sel = __shfl(logit, bi, 32);
    float lp  = ml + __logf(e) - logit_sel;       // -log_softmax(logit)[idx]
    float ent = lp * __expf(-lp);

    if (tid == 0) {
      out[t] = (float)bi;
      if (t == 0) { out[STEPS] = lp;  out[STEPS + 1] = ent; }
      else        { out[STEPS] += lp; out[STEPS + 1] += ent; }
      s_idx = bi;
    }
  }
  __syncthreads();

  const int idx = s_idx;
  const float* emb = w_emb + ((size_t)t * TYPES + idx) * HS;
  for (int k = tid; k < HS; k += 256) x_bf[k] = f2bf(emb[k]);
}

extern "C" void kernel_launch(void* const* d_in, const int* in_sizes, int n_in,
                              void* d_out, int out_size, void* d_ws, size_t ws_size,
                              hipStream_t stream) {
  (void)in_sizes; (void)n_in; (void)out_size; (void)ws_size;
  const float* g_emb  = (const float*)d_in[0];   // [1, 2048]
  const float* w_emb  = (const float*)d_in[1];   // [64, 32, 2048]
  const float* w_soft = (const float*)d_in[2];   // [2048, 32]
  const float* w_ih   = (const float*)d_in[3];   // [2, 8192, 2048]
  const float* w_hh   = (const float*)d_in[4];   // [2, 8192, 2048]
  const float* gumbel = (const float*)d_in[5];   // [64, 32]
  float* out = (float*)d_out;                    // 64 idx + lp + ent

  const size_t NW = (size_t)NLAY * G4H * HS;     // 33,554,432 per weight tensor

  // workspace layout: bf16 weights (fit in 192MB L2) + tiny state buffers
  bf16_t* wih_bf = (bf16_t*)d_ws;
  bf16_t* whh_bf = wih_bf + NW;
  bf16_t* x_bf   = whh_bf + NW;
  bf16_t* h0_bf  = x_bf + HS;
  bf16_t* h1_bf  = h0_bf + HS;
  float*  gates  = (float*)(h1_bf + HS);
  float*  c0     = gates + G4H;
  float*  c1     = c0 + HS;
  float*  h0f    = c1 + HS;
  float*  h1f    = h0f + HS;

  cvt_f32_to_bf16<<<2048, 256, 0, stream>>>(w_ih, wih_bf, (int)NW);
  cvt_f32_to_bf16<<<2048, 256, 0, stream>>>(w_hh, whh_bf, (int)NW);
  init_state<<<8, 256, 0, stream>>>(g_emb, x_bf, h0_bf, h1_bf, c0, c1);

  const size_t LOFF = (size_t)G4H * HS;          // layer stride in elements
  for (int t = 0; t < STEPS; ++t) {
    // layer 0
    lstm_matvec<<<128, 128, 0, stream>>>(wih_bf, whh_bf, x_bf, h0_bf, gates);
    lstm_cell<<<8, 256, 0, stream>>>(gates, c0, h0_bf, h0f);
    // layer 1 (input = new h0)
    lstm_matvec<<<128, 128, 0, stream>>>(wih_bf + LOFF, whh_bf + LOFF, h0_bf, h1_bf, gates);
    lstm_cell<<<8, 256, 0, stream>>>(gates, c1, h1_bf, h1f);
    // sample + next input
    sampler<<<1, 256, 0, stream>>>(h1f, w_soft, gumbel, w_emb, x_bf, out, t);
  }
}
